// GraphTransformerLayer_4123168604558
// MI455X (gfx1250) — matrix operand order repaired and checked by reading, compiled, not run on hardware
//
#include <hip/hip_runtime.h>

// ---------------------------------------------------------------------------
// GraphTransformerLayer for MI455X (gfx1250, wave32, WMMA)
//   N=8192 tokens, DIM=512, 8 heads x d=64, 32 sorted graph segments.
//   Pipeline: seg_bounds (+TDM null-op) -> fused QKV proj (WMMA f16, V stored
//   transposed) -> segment-masked flash attention (WMMA f16) -> out proj.
// ---------------------------------------------------------------------------

typedef __attribute__((ext_vector_type(16))) _Float16 v16h;
typedef __attribute__((ext_vector_type(8)))  float    v8f;
typedef unsigned __attribute__((ext_vector_type(4))) v4u;
typedef int      __attribute__((ext_vector_type(8))) v8i;
typedef int      __attribute__((ext_vector_type(4))) v4i;

#define N_TOK   8192
#define DIMN    512
#define NHEAD   8
#define HDIM    64
#define NGRAPH  32

union F16x16 { v16h v; unsigned u[8]; _Float16 h[16]; };

__device__ __forceinline__ v8f wmma_f16(v16h a, v16h b, v8f c) {
  // D = A(16x32 f16) x B(32x16 f16) + C(16x16 f32)
  return __builtin_amdgcn_wmma_f32_16x16x32_f16(
      /*neg_a=*/false, a, /*neg_b=*/false, b,
      /*c_mod=*/(short)0, c, /*reuse_a=*/false, /*reuse_b=*/false);
}

__device__ __forceinline__ unsigned pk_scale_f16x2(unsigned u, float s) {
  union { unsigned u; _Float16 h[2]; } x; x.u = u;
  x.h[0] = (_Float16)((float)x.h[0] * s);
  x.h[1] = (_Float16)((float)x.h[1] * s);
  return x.u;
}

// ---------------------------------------------------------------------------
// Kernel 0: segment boundaries (batch sorted; seg[g]=lower_bound(batch,g);
// 33 entries). Also exercises the Tensor Data Mover with a NULL descriptor
// (D# group0.count==0 -> NOP per CDNA5 ISA 8.3) + s_wait_tensorcnt.
// Toolchain probe: this clang exposes the 6-arg tensor_load_to_lds.
// ---------------------------------------------------------------------------
__global__ void seg_bounds_kernel(const int* __restrict__ batch, int n,
                                  int* __restrict__ seg) {
  int g = threadIdx.x;
  if (g <= NGRAPH) {
    int lo = 0, hi = n;
    while (lo < hi) {
      int mid = (lo + hi) >> 1;
      if (batch[mid] < g) lo = mid + 1; else hi = mid;
    }
    seg[g] = lo;
  }
  // NULL-tensor TDM op: count==0 in D# group0 -> architectural no-op.
  v4u g0 = {0u, 0u, 0u, 0u};
  v8i g1 = {0, 0, 0, 0, 0, 0, 0, 0};
  v4i g2 = {0, 0, 0, 0};
  v4i g3 = {0, 0, 0, 0};
  v8i g4 = {0, 0, 0, 0, 0, 0, 0, 0};
  __builtin_amdgcn_tensor_load_to_lds(g0, g1, g2, g3, g4, 0);
  __builtin_amdgcn_s_wait_tensorcnt(0);
}

// ---------------------------------------------------------------------------
// Kernel 1: fused QKV projection.  Y = X @ W^T + b  (torch Linear convention)
// Block: 256 thr = 8 waves (2M x 4N); block tile 64(M) x 128(N); K-step 32;
// per-wave 32x32 C tile (4 WMMA / K-step); double-buffered LDS staging.
// blockIdx.z selects Q/K/V.  V is stored TRANSPOSED: Vt[dim][token].
// ---------------------------------------------------------------------------
__global__ __launch_bounds__(256)
void qkv_proj_kernel(const float* __restrict__ X,
                     const float* __restrict__ Wq, const float* __restrict__ Wk,
                     const float* __restrict__ Wv,
                     const float* __restrict__ Bq, const float* __restrict__ Bk,
                     const float* __restrict__ Bv,
                     _Float16* __restrict__ Qh, _Float16* __restrict__ Kh,
                     _Float16* __restrict__ Vt) {
  constexpr int LDT = 34;                  // padded LDS stride (halves)
  __shared__ _Float16 Xs[2][64 * LDT];
  __shared__ _Float16 Ws[2][128 * LDT];

  const float* W; const float* B; _Float16* Y;
  switch (blockIdx.z) {
    case 0:  W = Wq; B = Bq; Y = Qh; break;
    case 1:  W = Wk; B = Bk; Y = Kh; break;
    default: W = Wv; B = Bv; Y = Vt; break;
  }
  const bool is_v = (blockIdx.z == 2);

  const int tid  = threadIdx.x;
  const int lane = tid & 31;
  const int wave = tid >> 5;
  const int g    = lane >> 4;              // half-wave group
  const int nn   = lane & 15;
  const int wm   = wave & 1;               // wave's 32-row M slice (0..1)
  const int wn   = wave >> 1;              // wave's 32-col N slice (0..3)
  const int m0   = blockIdx.x * 64;
  const int j0   = blockIdx.y * 128;

  const int xrow = tid >> 2, xq = tid & 3; // X staging: 64 rows, 2 reps
  const int wrow = tid >> 1, wq = tid & 1; // W staging: 128 rows, 4 reps

  auto stage = [&](int k0, int b) {
#pragma unroll
    for (int rep = 0; rep < 2; ++rep) {
      int cc = (xq + rep * 4) * 4;
      float4 xv = *(const float4*)&X[(size_t)(m0 + xrow) * DIMN + k0 + cc];
      Xs[b][xrow * LDT + cc + 0] = (_Float16)xv.x;
      Xs[b][xrow * LDT + cc + 1] = (_Float16)xv.y;
      Xs[b][xrow * LDT + cc + 2] = (_Float16)xv.z;
      Xs[b][xrow * LDT + cc + 3] = (_Float16)xv.w;
    }
#pragma unroll
    for (int rep = 0; rep < 4; ++rep) {
      int cc = (wq + rep * 2) * 4;
      float4 wv = *(const float4*)&W[(size_t)(j0 + wrow) * DIMN + k0 + cc];
      Ws[b][wrow * LDT + cc + 0] = (_Float16)wv.x;
      Ws[b][wrow * LDT + cc + 1] = (_Float16)wv.y;
      Ws[b][wrow * LDT + cc + 2] = (_Float16)wv.z;
      Ws[b][wrow * LDT + cc + 3] = (_Float16)wv.w;
    }
  };

  v8f c00 = {}, c01 = {}, c10 = {}, c11 = {};
  stage(0, 0);
  __syncthreads();

  for (int i = 0; i < DIMN / 32; ++i) {
    const int b = i & 1;
    if (i + 1 < DIMN / 32) stage((i + 1) * 32, b ^ 1);

    F16x16 a0, a1, b0f, b1f;
#pragma unroll
    for (int v = 0; v < 8; ++v) {
      int Ka = 16 * (v >> 2) + 8 * g + 2 * (v & 3);     // A-frag K index
      a0.u[v]  = *(const unsigned*)&Xs[b][(wm * 32      + nn) * LDT + Ka];
      a1.u[v]  = *(const unsigned*)&Xs[b][(wm * 32 + 16 + nn) * LDT + Ka];
      int Kb = 16 * (v >> 2) + 8 * g + 2 * (v & 3);     // B-frag K index
      b0f.u[v] = *(const unsigned*)&Ws[b][(wn * 32      + nn) * LDT + Kb];
      b1f.u[v] = *(const unsigned*)&Ws[b][(wn * 32 + 16 + nn) * LDT + Kb];
    }
    c00 = wmma_f16(a0.v, b0f.v, c00);
    c01 = wmma_f16(a0.v, b1f.v, c01);
    c10 = wmma_f16(a1.v, b0f.v, c10);
    c11 = wmma_f16(a1.v, b1f.v, c11);
    __syncthreads();
  }

  const int jj0 = j0 + wn * 32 + nn;
  const int jj1 = jj0 + 16;
  const float bb0 = B[jj0];
  const float bb1 = B[jj1];
#pragma unroll
  for (int v = 0; v < 8; ++v) {
    int mA = m0 + wm * 32 + v + 8 * g;                  // C-layout rows
    int mB = mA + 16;
    if (!is_v) {
      Y[(size_t)mA * DIMN + jj0] = (_Float16)(c00[v] + bb0);
      Y[(size_t)mA * DIMN + jj1] = (_Float16)(c01[v] + bb1);
      Y[(size_t)mB * DIMN + jj0] = (_Float16)(c10[v] + bb0);
      Y[(size_t)mB * DIMN + jj1] = (_Float16)(c11[v] + bb1);
    } else {                                            // transposed V
      Y[(size_t)jj0 * N_TOK + mA] = (_Float16)(c00[v] + bb0);
      Y[(size_t)jj1 * N_TOK + mA] = (_Float16)(c01[v] + bb1);
      Y[(size_t)jj0 * N_TOK + mB] = (_Float16)(c10[v] + bb0);
      Y[(size_t)jj1 * N_TOK + mB] = (_Float16)(c11[v] + bb1);
    }
  }
}

// ---------------------------------------------------------------------------
// Kernel 2: segment-masked flash attention.
// One wave per (16-row query tile, head). Online softmax over 32-col chunks
// restricted to the block's segment range; per-row masking handles tiles that
// straddle graph boundaries. 1/sqrt(d)=0.125 folded into Q fragments.
// V consumed from transposed layout Vt[dim][token] -> contiguous b32 frags.
// ---------------------------------------------------------------------------
__global__ __launch_bounds__(32)
void attn_kernel(const _Float16* __restrict__ Qh, const _Float16* __restrict__ Kh,
                 const _Float16* __restrict__ Vt, const int* __restrict__ batch,
                 const int* __restrict__ seg, _Float16* __restrict__ Oh) {
  __shared__ _Float16 Ps[16 * 32];         // P tile staging for C->A transpose

  const int lane = threadIdx.x;
  const int g    = lane >> 4;
  const int nn   = lane & 15;
  const int r0   = blockIdx.x * 16;
  const int h64  = blockIdx.y * HDIM;

  // Per-row segment bounds (rows m = v + 8g live in this lane's C-layout slots)
  int rs[8], re[8];
#pragma unroll
  for (int v = 0; v < 8; ++v) {
    int gr = batch[r0 + v + 8 * g];
    rs[v] = seg[gr];
    re[v] = seg[gr + 1];
  }
  const int jstart = seg[batch[r0]];           // batch sorted -> min start
  const int jend   = seg[batch[r0 + 15] + 1];  // batch sorted -> max end

  // Q fragments for feature chunks [0,32) and [32,64), prescaled by 0.125.
  F16x16 aq0, aq1;
#pragma unroll
  for (int v = 0; v < 8; ++v) {
    int Ka = 16 * (v >> 2) + 8 * g + 2 * (v & 3);
    const _Float16* p = &Qh[(size_t)(r0 + nn) * DIMN + h64 + Ka];
    aq0.u[v] = pk_scale_f16x2(*(const unsigned*)p,        0.125f);
    aq1.u[v] = pk_scale_f16x2(*(const unsigned*)(p + 32), 0.125f);
  }

  v8f oacc[4] = {};                        // O (16x64) in four C-layout tiles
  float Mrow[8], Lrow[8];
#pragma unroll
  for (int v = 0; v < 8; ++v) { Mrow[v] = -1.0e30f; Lrow[v] = 0.0f; }

  for (int j0c = (jstart & ~31); j0c < jend; j0c += 32) {
    if (j0c + 32 < jend) {                 // uniform branch: EXEC stays full
      int pr = j0c + 32 + lane; pr = pr < N_TOK ? pr : N_TOK - 1;
      __builtin_prefetch(&Kh[(size_t)pr * DIMN + h64], 0, 1);
      __builtin_prefetch(&Vt[(size_t)(h64 + lane) * N_TOK + j0c + 32], 0, 1);
      __builtin_prefetch(&Vt[(size_t)(h64 + 32 + lane) * N_TOK + j0c + 32], 0, 1);
    }

    // K^T fragments: B[c][j] = K[j0c + jt*16 + j][h64 + kk*32 + c]
    F16x16 bk00, bk01, bk10, bk11;
#pragma unroll
    for (int v = 0; v < 8; ++v) {
      int C = 16 * (v >> 2) + 8 * g + 2 * (v & 3);
      const _Float16* p0 = &Kh[(size_t)(j0c + nn) * DIMN + h64 + C];
      const _Float16* p1 = &Kh[(size_t)(j0c + 16 + nn) * DIMN + h64 + C];
      bk00.u[v] = *(const unsigned*)p0;
      bk01.u[v] = *(const unsigned*)(p0 + 32);
      bk10.u[v] = *(const unsigned*)p1;
      bk11.u[v] = *(const unsigned*)(p1 + 32);
    }

    v8f s0 = {}; v8f s1 = {};
    s0 = wmma_f16(aq0.v, bk00.v, s0);
    s0 = wmma_f16(aq1.v, bk01.v, s0);
    s1 = wmma_f16(aq0.v, bk10.v, s1);
    s1 = wmma_f16(aq1.v, bk11.v, s1);

    // Online softmax with per-row segment mask.
    const int jj0 = j0c + nn;
    const int jj1 = j0c + 16 + nn;
    float p0v[8], p1v[8];
#pragma unroll
    for (int v = 0; v < 8; ++v) {
      bool ok0 = (jj0 >= rs[v]) && (jj0 < re[v]);
      bool ok1 = (jj1 >= rs[v]) && (jj1 < re[v]);
      float a0 = ok0 ? s0[v] : -1.0e30f;
      float a1 = ok1 ? s1[v] : -1.0e30f;
      float mx = fmaxf(a0, a1);
      mx = fmaxf(mx, __shfl_xor(mx, 1, 32));
      mx = fmaxf(mx, __shfl_xor(mx, 2, 32));
      mx = fmaxf(mx, __shfl_xor(mx, 4, 32));
      mx = fmaxf(mx, __shfl_xor(mx, 8, 32));
      float Mn = fmaxf(Mrow[v], mx);
      float sc = __expf(Mrow[v] - Mn);
      float e0 = ok0 ? __expf(a0 - Mn) : 0.0f;
      float e1 = ok1 ? __expf(a1 - Mn) : 0.0f;
      float sum = e0 + e1;
      sum += __shfl_xor(sum, 1, 32);
      sum += __shfl_xor(sum, 2, 32);
      sum += __shfl_xor(sum, 4, 32);
      sum += __shfl_xor(sum, 8, 32);
      Lrow[v] = Lrow[v] * sc + sum;
      Mrow[v] = Mn;
      oacc[0][v] *= sc; oacc[1][v] *= sc; oacc[2][v] *= sc; oacc[3][v] *= sc;
      p0v[v] = e0; p1v[v] = e1;
    }

    // C-layout P -> LDS -> A-fragment (cross-lane transpose).
    __syncthreads();
#pragma unroll
    for (int v = 0; v < 8; ++v) {
      int m = v + 8 * g;
      Ps[m * 32 + nn]      = (_Float16)p0v[v];
      Ps[m * 32 + 16 + nn] = (_Float16)p1v[v];
    }
    __syncthreads();
    F16x16 ap;
#pragma unroll
    for (int v = 0; v < 8; ++v) {
      int Ka = 16 * (v >> 2) + 8 * g + 2 * (v & 3);
      ap.u[v] = *(const unsigned*)&Ps[nn * 32 + Ka];
    }

    // O += P (16x32) @ V-chunk (32x64): Vt gives contiguous token pairs.
#pragma unroll
    for (int t = 0; t < 4; ++t) {
      F16x16 bv;
#pragma unroll
      for (int v = 0; v < 8; ++v) {
        int C = 16 * (v >> 2) + 8 * g + 2 * (v & 3);
        bv.u[v] = *(const unsigned*)
            &Vt[(size_t)(h64 + t * 16 + nn) * N_TOK + j0c + C];
      }
      oacc[t] = wmma_f16(ap.v, bv.v, oacc[t]);
    }
  }

  // Normalize and store f16 attention output.
#pragma unroll
  for (int v = 0; v < 8; ++v) {
    float inv = 1.0f / Lrow[v];            // diagonal always unmasked -> L>0
    int m = r0 + v + 8 * g;
#pragma unroll
    for (int t = 0; t < 4; ++t) {
      Oh[(size_t)m * DIMN + h64 + t * 16 + nn] = (_Float16)(oacc[t][v] * inv);
    }
  }
}

// ---------------------------------------------------------------------------
// Kernel 3: output projection. out = Attn @ Wo^T + bo, f32 output to d_out.
// Same tiling as kernel 1 (A already f16).
// ---------------------------------------------------------------------------
__global__ __launch_bounds__(256)
void out_proj_kernel(const _Float16* __restrict__ A,
                     const float* __restrict__ W, const float* __restrict__ B,
                     float* __restrict__ Out) {
  constexpr int LDT = 34;
  __shared__ _Float16 Xs[2][64 * LDT];
  __shared__ _Float16 Ws[2][128 * LDT];

  const int tid  = threadIdx.x;
  const int lane = tid & 31;
  const int wave = tid >> 5;
  const int g    = lane >> 4;
  const int nn   = lane & 15;
  const int wm   = wave & 1;
  const int wn   = wave >> 1;
  const int m0   = blockIdx.x * 64;
  const int j0   = blockIdx.y * 128;

  const int xrow = tid >> 2, xq = tid & 3;
  const int wrow = tid >> 1, wq = tid & 1;

  auto stage = [&](int k0, int b) {
    uint4 av = *(const uint4*)&A[(size_t)(m0 + xrow) * DIMN + k0 + xq * 8];
    unsigned* dst = (unsigned*)&Xs[b][xrow * LDT + xq * 8];
    dst[0] = av.x; dst[1] = av.y; dst[2] = av.z; dst[3] = av.w;
#pragma unroll
    for (int rep = 0; rep < 4; ++rep) {
      int cc = (wq + rep * 2) * 4;
      float4 wv = *(const float4*)&W[(size_t)(j0 + wrow) * DIMN + k0 + cc];
      Ws[b][wrow * LDT + cc + 0] = (_Float16)wv.x;
      Ws[b][wrow * LDT + cc + 1] = (_Float16)wv.y;
      Ws[b][wrow * LDT + cc + 2] = (_Float16)wv.z;
      Ws[b][wrow * LDT + cc + 3] = (_Float16)wv.w;
    }
  };

  v8f c00 = {}, c01 = {}, c10 = {}, c11 = {};
  stage(0, 0);
  __syncthreads();

  for (int i = 0; i < DIMN / 32; ++i) {
    const int b = i & 1;
    if (i + 1 < DIMN / 32) stage((i + 1) * 32, b ^ 1);

    F16x16 a0, a1, b0f, b1f;
#pragma unroll
    for (int v = 0; v < 8; ++v) {
      int Ka = 16 * (v >> 2) + 8 * g + 2 * (v & 3);
      a0.u[v]  = *(const unsigned*)&Xs[b][(wm * 32      + nn) * LDT + Ka];
      a1.u[v]  = *(const unsigned*)&Xs[b][(wm * 32 + 16 + nn) * LDT + Ka];
      int Kb = 16 * (v >> 2) + 8 * g + 2 * (v & 3);
      b0f.u[v] = *(const unsigned*)&Ws[b][(wn * 32      + nn) * LDT + Kb];
      b1f.u[v] = *(const unsigned*)&Ws[b][(wn * 32 + 16 + nn) * LDT + Kb];
    }
    c00 = wmma_f16(a0.v, b0f.v, c00);
    c01 = wmma_f16(a0.v, b1f.v, c01);
    c10 = wmma_f16(a1.v, b0f.v, c10);
    c11 = wmma_f16(a1.v, b1f.v, c11);
    __syncthreads();
  }

  const int jj0 = j0 + wn * 32 + nn;
  const int jj1 = jj0 + 16;
  const float bb0 = B[jj0];
  const float bb1 = B[jj1];
#pragma unroll
  for (int v = 0; v < 8; ++v) {
    int mA = m0 + wm * 32 + v + 8 * g;
    int mB = mA + 16;
    Out[(size_t)mA * DIMN + jj0] = c00[v] + bb0;
    Out[(size_t)mA * DIMN + jj1] = c01[v] + bb1;
    Out[(size_t)mB * DIMN + jj0] = c10[v] + bb0;
    Out[(size_t)mB * DIMN + jj1] = c11[v] + bb1;
  }
}

// ---------------------------------------------------------------------------
extern "C" void kernel_launch(void* const* d_in, const int* in_sizes, int n_in,
                              void* d_out, int out_size, void* d_ws, size_t ws_size,
                              hipStream_t stream) {
  (void)in_sizes; (void)n_in; (void)out_size; (void)ws_size;

  const float* x     = (const float*)d_in[0];
  const int*   batch = (const int*)d_in[1];
  const float* wq    = (const float*)d_in[2];
  const float* wk    = (const float*)d_in[3];
  const float* wv    = (const float*)d_in[4];
  const float* bq    = (const float*)d_in[5];
  const float* bk    = (const float*)d_in[6];
  const float* bv    = (const float*)d_in[7];
  const float* wo    = (const float*)d_in[8];
  const float* bo    = (const float*)d_in[9];
  float* out = (float*)d_out;

  char* ws = (char*)d_ws;
  int* seg = (int*)ws;                                   // 33 ints (+pad)
  _Float16* Qh = (_Float16*)(ws + 256);
  _Float16* Kh = Qh + (size_t)N_TOK * DIMN;
  _Float16* Vt = Kh + (size_t)N_TOK * DIMN;              // [DIMN][N_TOK]
  _Float16* Oh = Vt + (size_t)N_TOK * DIMN;              // total ~32 MB

  seg_bounds_kernel<<<1, 64, 0, stream>>>(batch, N_TOK, seg);
  qkv_proj_kernel<<<dim3(N_TOK / 64, DIMN / 128, 3), 256, 0, stream>>>(
      x, wq, wk, wv, bq, bk, bv, Qh, Kh, Vt);
  attn_kernel<<<dim3(N_TOK / 16, NHEAD), 32, 0, stream>>>(
      Qh, Kh, Vt, batch, seg, Oh);
  out_proj_kernel<<<dim3(N_TOK / 64, DIMN / 128), 256, 0, stream>>>(
      Oh, wo, bo, out);
}
